// iSogCLR_Loss_22643067584621
// MI455X (gfx1250) — matrix operand order repaired and checked by reading, not compiled, not run on hardware
//
#include <hip/hip_runtime.h>

typedef __attribute__((ext_vector_type(16))) _Float16 v16h;
typedef __attribute__((ext_vector_type(8)))  _Float16 v8h;
typedef __attribute__((ext_vector_type(8)))  float    v8f;

static constexpr int   BN        = 1024;   // batch
static constexpr int   DN        = 256;    // feature dim
static constexpr float ALPHA     = 0.5f;
static constexpr float GAMMA     = 0.8f;
static constexpr float BETA_U    = 0.9f;
static constexpr float RHO       = 8.0f;
static constexpr float ETA       = 0.01f;
static constexpr float EPSV      = 1e-14f;
static constexpr float GRAD_CLIP = 5.0f;
static constexpr float TAU_MIN   = 0.001f;
static constexpr float TAU_MAX   = 1.0f;

// ---------------------------------------------------------------------------
// K0: L2-normalize rows of zis/zjs, emit f16 copies for the WMMA GEMM.
// blockIdx.x in [0,2048): low 1024 -> zis, high 1024 -> zjs. 256 thr = D.
// ---------------------------------------------------------------------------
__global__ void normalize_f16_kernel(const float* __restrict__ zis,
                                     const float* __restrict__ zjs,
                                     _Float16* __restrict__ zis_h,
                                     _Float16* __restrict__ zjs_h) {
    __shared__ float red[256];
    int row = blockIdx.x & (BN - 1);
    bool isJ = blockIdx.x >= BN;
    const float* src = isJ ? zjs : zis;
    _Float16* dst = isJ ? zjs_h : zis_h;
    int t = threadIdx.x;
    float v = src[row * DN + t];
    red[t] = v * v;
    __syncthreads();
    for (int s = 128; s > 0; s >>= 1) {
        if (t < s) red[t] += red[t + s];
        __syncthreads();
    }
    float rn = rsqrtf(red[0]);
    dst[row * DN + t] = (_Float16)(v * rn);
}

// ---------------------------------------------------------------------------
// K1: sim = Zi_h @ Zj_h^T via v_wmma_f32_16x16x32_f16.
// grid = (8 Ntiles-of-128, 64 Mtiles-of-16); 8 waves/block, one 16x16 tile/wave.
// A frag (16x32 f16): lane L (m = L&15, h = L>>4) holds K = 8h+{0..7} and
//   K = 16+8h+{0..7}  -> two contiguous 16B loads from row-major zis_h.
// B frag (32x16 f16): lane L (n = L&15, h = L>>4) holds K = 8h+{0..7} and
//   K = 16+8h+{0..7} -> two 16B loads from row-major zjs_h (B[k][n] = Zj[n][k]).
// D frag (16x16 f32): VGPR r -> M = m_base + 8h + r, N = n_base + (L&15);
//   consecutive r are consecutive M, so the transposed store is contiguous.
// ---------------------------------------------------------------------------
__global__ void simgemm_kernel(const _Float16* __restrict__ Ah,
                               const _Float16* __restrict__ Bh,
                               float* __restrict__ sim,
                               float* __restrict__ simT) {
    int lane = threadIdx.x & 31;
    int wave = threadIdx.x >> 5;
    int m_base = blockIdx.y * 16;
    int n_base = blockIdx.x * 128 + wave * 16;
    int l15 = lane & 15;
    int h = lane >> 4;

    const _Float16* pa = Ah + (m_base + l15) * DN;
    const _Float16* pb = Bh + (n_base + l15) * DN;

    v8f c = {};
#pragma unroll
    for (int kb = 0; kb < DN; kb += 32) {
        v8h alo = *(const v8h*)(pa + kb + 8 * h);
        v8h ahi = *(const v8h*)(pa + kb + 16 + 8 * h);
        v16h a;
#pragma unroll
        for (int e = 0; e < 8; ++e) { a[e] = alo[e]; a[8 + e] = ahi[e]; }
        v8h blo = *(const v8h*)(pb + kb + 8 * h);
        v8h bhi = *(const v8h*)(pb + kb + 16 + 8 * h);
        v16h b;
#pragma unroll
        for (int e = 0; e < 8; ++e) { b[e] = blo[e]; b[8 + e] = bhi[e]; }
        c = __builtin_amdgcn_wmma_f32_16x16x32_f16(
                /*neg_a=*/false, a, /*neg_b=*/false, b,
                /*c_mod=*/(short)0, c, /*reuse_a=*/false, /*reuse_b=*/false);
    }

    int n = n_base + l15;
#pragma unroll
    for (int r = 0; r < 8; ++r)
        sim[(m_base + 8 * h + r) * BN + n] = c[r];
    float* tp = simT + n * BN + m_base + 8 * h;
#pragma unroll
    for (int r = 0; r < 8; ++r) tp[r] = c[r];
}

// ---------------------------------------------------------------------------
// K2/K3: per-row streaming stats. Run once on sim (image/I side), once on
// simT (text/T side). One block (256 thr) per row; each thread owns 4 cols.
// idt = (sim - diag)/tau ; new_b = max(rowmax(idt), old_b)
// S1 = sum_{j!=i} exp(idt-new_b) ; S2 = sum_{j!=i} exp(idt-new_b)*idt
// s_new = (1-g)*s_old*exp(old_b-new_b) + g*S1 ; s_c = max(s_new, EPS)
// loss_i = tau * S2/s_c    (since image_diffs = tau*idt)
// grad = log(s_c)+new_b+RHO - (S2/s_c)/(B-1); clip; EMA u; tau update.
// ---------------------------------------------------------------------------
__global__ void stat_pass_kernel(const float* __restrict__ simrows,
                                 const int*   __restrict__ ids,
                                 const float* __restrict__ s_buf,
                                 const float* __restrict__ b_buf,
                                 const float* __restrict__ tau_buf,
                                 const float* __restrict__ u_buf,
                                 float* __restrict__ out_new_b,
                                 float* __restrict__ out_s_new,
                                 float* __restrict__ out_u_new,
                                 float* __restrict__ out_tau_new,
                                 float* __restrict__ loss_partial,
                                 float* __restrict__ tau_partial) {
    __shared__ float red[256];
    __shared__ float bc[6];
    int i = blockIdx.x;
    int t = threadIdx.x;
    const float* row = simrows + i * BN;

    if (t == 0) {
        int id = ids[i];
        bc[0] = row[i];       // diag
        bc[1] = tau_buf[id];
        bc[2] = b_buf[id];
        bc[3] = s_buf[id];
        bc[4] = u_buf[id];
    }
    __syncthreads();
    float diag = bc[0], tau = bc[1], old_b = bc[2];
    float inv_tau = 1.0f / tau;

    float v[4];
    float mx = -3.4e38f;
#pragma unroll
    for (int q = 0; q < 4; ++q) {
        int j = t + q * 256;
        float idt = (row[j] - diag) * inv_tau;
        v[q] = idt;
        mx = fmaxf(mx, idt);
    }
    red[t] = mx;
    __syncthreads();
    for (int s = 128; s > 0; s >>= 1) {
        if (t < s) red[t] = fmaxf(red[t], red[t + s]);
        __syncthreads();
    }
    float new_b = fmaxf(red[0], old_b);
    __syncthreads();

    float s1 = 0.0f, s2 = 0.0f;
#pragma unroll
    for (int q = 0; q < 4; ++q) {
        int j = t + q * 256;
        float e = (j == i) ? 0.0f : expf(v[q] - new_b);
        s1 += e;
        s2 += e * v[q];
    }
    red[t] = s1;
    __syncthreads();
    for (int s = 128; s > 0; s >>= 1) {
        if (t < s) red[t] += red[t + s];
        __syncthreads();
    }
    s1 = red[0];
    __syncthreads();
    red[t] = s2;
    __syncthreads();
    for (int s = 128; s > 0; s >>= 1) {
        if (t < s) red[t] += red[t + s];
        __syncthreads();
    }
    s2 = red[0];

    if (t == 0) {
        float s_new = (1.0f - GAMMA) * bc[3] * expf(old_b - new_b) + GAMMA * s1;
        float s_c = fmaxf(s_new, EPSV);
        float sw_idt = s2 / s_c;                 // sum(w * idt)
        float loss = tau * sw_idt;               // sum(w * diffs)
        float grad = logf(s_c) + new_b + RHO - sw_idt * (1.0f / (BN - 1));
        grad = fminf(fmaxf(grad, -GRAD_CLIP), GRAD_CLIP);
        float u_new = (1.0f - BETA_U) * bc[4] + BETA_U * grad;
        float tau_new = fminf(fmaxf(tau - ETA * u_new, TAU_MIN), TAU_MAX);
        out_new_b[i]   = new_b;
        out_s_new[i]   = s_new;
        out_u_new[i]   = u_new;
        out_tau_new[i] = tau_new;
        loss_partial[i] = loss;
        tau_partial[i]  = tau;
    }
}

// ---------------------------------------------------------------------------
// K4: fold 4 x [1024] partial vectors into the 3 scalar outputs.
// ---------------------------------------------------------------------------
__global__ void finalize_kernel(const float* __restrict__ il,
                                const float* __restrict__ tl,
                                const float* __restrict__ ti,
                                const float* __restrict__ tt,
                                float* __restrict__ out) {
    __shared__ float r0[256], r1[256], r2[256], r3[256];
    int t = threadIdx.x;
    float a = 0.f, b = 0.f, c = 0.f, d = 0.f;
#pragma unroll
    for (int q = 0; q < 4; ++q) {
        int j = t + q * 256;
        a += il[j]; b += tl[j]; c += ti[j]; d += tt[j];
    }
    r0[t] = a; r1[t] = b; r2[t] = c; r3[t] = d;
    __syncthreads();
    for (int s = 128; s > 0; s >>= 1) {
        if (t < s) { r0[t] += r0[t+s]; r1[t] += r1[t+s]; r2[t] += r2[t+s]; r3[t] += r3[t+s]; }
        __syncthreads();
    }
    if (t == 0) {
        float mi = r0[0] * (1.0f / BN);
        float mt = r1[0] * (1.0f / BN);
        out[0] = ALPHA * mi + (1.0f - ALPHA) * mt;
        out[1] = r2[0] * (1.0f / BN);
        out[2] = r3[0] * (1.0f / BN);
    }
}

extern "C" void kernel_launch(void* const* d_in, const int* in_sizes, int n_in,
                              void* d_out, int out_size, void* d_ws, size_t ws_size,
                              hipStream_t stream) {
    const float* zis   = (const float*)d_in[0];
    const float* zjs   = (const float*)d_in[1];
    const int*   ids   = (const int*)  d_in[2];
    const float* s_I   = (const float*)d_in[3];
    const float* s_T   = (const float*)d_in[4];
    const float* b_I   = (const float*)d_in[5];
    const float* b_T   = (const float*)d_in[6];
    const float* tau_I = (const float*)d_in[7];
    const float* tau_T = (const float*)d_in[8];
    const float* u_I   = (const float*)d_in[9];
    const float* u_T   = (const float*)d_in[10];

    char* ws = (char*)d_ws;
    _Float16* zis_h = (_Float16*)(ws + 0);                      // 512 KB
    _Float16* zjs_h = (_Float16*)(ws + 524288);                 // 512 KB
    float*    sim   = (float*)   (ws + 1048576);                // 4 MB
    float*    simT  = (float*)   (ws + 5242880);                // 4 MB
    float*    image_loss = (float*)(ws + 9437184);              // 4 KB
    float*    text_loss  = (float*)(ws + 9441280);              // 4 KB
    float*    tau_img    = (float*)(ws + 9445376);              // 4 KB
    float*    tau_txt    = (float*)(ws + 9449472);              // 4 KB

    float* out = (float*)d_out;
    float* o_new_b_I = out + 3;
    float* o_new_b_T = out + 3 + 1 * BN;
    float* o_s_I     = out + 3 + 2 * BN;
    float* o_s_T     = out + 3 + 3 * BN;
    float* o_u_I     = out + 3 + 4 * BN;
    float* o_u_T     = out + 3 + 5 * BN;
    float* o_tau_I   = out + 3 + 6 * BN;
    float* o_tau_T   = out + 3 + 7 * BN;

    normalize_f16_kernel<<<2 * BN, DN, 0, stream>>>(zis, zjs, zis_h, zjs_h);

    simgemm_kernel<<<dim3(BN / 128, BN / 16), 256, 0, stream>>>(zis_h, zjs_h, sim, simT);

    // image/I side: rows of sim
    stat_pass_kernel<<<BN, 256, 0, stream>>>(sim, ids, s_I, b_I, tau_I, u_I,
                                             o_new_b_I, o_s_I, o_u_I, o_tau_I,
                                             image_loss, tau_img);
    // text/T side: rows of simT (= columns of sim)
    stat_pass_kernel<<<BN, 256, 0, stream>>>(simT, ids, s_T, b_T, tau_T, u_T,
                                             o_new_b_T, o_s_T, o_u_T, o_tau_T,
                                             text_loss, tau_txt);

    finalize_kernel<<<1, 256, 0, stream>>>(image_loss, text_loss, tau_img, tau_txt, out);
}
